// MultiHeadedAttention_54494545052047
// MI455X (gfx1250) — matrix operand order repaired and checked by reading, not compiled, not run on hardware
//
#include <hip/hip_runtime.h>
#include <hip/hip_bf16.h>
#include <stdint.h>

#define NUM_HEADS 16
#define HEAD_SIZE 64
#define DMODEL    1024
#define SEQLEN    2048
#define BATCH     2

typedef __attribute__((ext_vector_type(16))) __bf16 v16bf;
typedef __attribute__((ext_vector_type(8)))  __bf16 bf16x8;
typedef __attribute__((ext_vector_type(4)))  __bf16 bf16x4;
typedef __attribute__((ext_vector_type(8)))  float  v8f;
typedef __attribute__((ext_vector_type(4)))  float  f32x4;

// ---------------- CDNA5 async global->LDS copy (probe-guarded) ----------------
#if defined(__has_builtin)
#if __has_builtin(__builtin_amdgcn_global_load_async_to_lds_b128)
#define USE_ASYNC_LDS 1
#endif
#endif

// Builtin signature (from hipcc diagnostic): takes pointers to 16-byte int
// vectors in the global(1) / LDS(3) address spaces, plus imm offset + cpol.
typedef int v4i_vec __attribute__((vector_size(16)));
typedef __attribute__((address_space(1))) v4i_vec gvec4i;
typedef __attribute__((address_space(3))) v4i_vec lvec4i;

static __device__ inline void async_copy16(void* ldsDst, const void* gSrc) {
#ifdef USE_ASYNC_LDS
  __builtin_amdgcn_global_load_async_to_lds_b128((gvec4i*)gSrc, (lvec4i*)ldsDst, 0, 0);
#else
  *(bf16x8*)ldsDst = *(const bf16x8*)gSrc;
#endif
}
static __device__ inline void async_wait() {
#ifdef USE_ASYNC_LDS
#if __has_builtin(__builtin_amdgcn_s_wait_asynccnt)
  __builtin_amdgcn_s_wait_asynccnt(0);
#else
  asm volatile("s_wait_asynccnt 0" ::: "memory");
#endif
#endif
}

// ---- fragment gathers for CDNA5 WMMA 16-bit operands, contiguous k = 16h + s in A and B ----
// A (16x32, MxK): lanes 0-15 -> M=lane, elems 0..15 = K kb..kb+15 (kb=0)
//                 lanes16-31 -> M=lane-16, kb=16
static __device__ inline v16bf frag_a(const __bf16* row, int kb) {
  bf16x8 lo = *(const bf16x8*)(row + kb);
  bf16x8 hi = *(const bf16x8*)(row + kb + 8);
  return __builtin_shufflevector(lo, hi, 0,1,2,3,4,5,6,7,8,9,10,11,12,13,14,15);
}
// B (32x16, KxN) gathered from LDS stored as Bt[n][k]:
// lanes 0-15 -> N=lane, K = 0..15 ; lanes 16-31 -> N=lane-16, K = 16..31
static __device__ inline v16bf frag_b(const __bf16* row, int kb) {
  bf16x8 lo = *(const bf16x8*)(row + kb);
  bf16x8 hi = *(const bf16x8*)(row + kb + 8);
  return __builtin_shufflevector(lo, hi, 0,1,2,3,4,5,6,7,8,9,10,11,12,13,14,15);
}

static __device__ inline v8f wmma_bf16(v16bf a, v16bf b, v8f c) {
  return __builtin_amdgcn_wmma_f32_16x16x32_bf16(false, a, false, b, (short)0, c, false, false);
}

// =====================================================================
// GEMM: out[64x64 tile] = A[M x 1024] * W[1024 x 1024] + bias, optional scale.
// A_BF16:   A is bf16 (ctx, async-copied) vs f32 (raw q/k/v, converted).
// SPLIT_OUT: write bf16 head-split [B][H][L][64]; else write f32 [M][D].
// =====================================================================
template<bool A_BF16, bool SPLIT_OUT>
__global__ __launch_bounds__(128) void gemm_proj_kernel(
    const void* __restrict__ Av, const float* __restrict__ W,
    const float* __restrict__ bias, float scale, void* __restrict__ outv)
{
  __shared__ __attribute__((aligned(16))) __bf16 sA[64][40];  // [m][k], pad 32->40
  __shared__ __attribute__((aligned(16))) __bf16 sB[64][40];  // [n][k] (W transposed)

  const int tid  = threadIdx.x;
  const int lane = tid & 31;
  const int wave = tid >> 5;
  const int lh   = lane & 15;
  const int hi16 = (lane >> 4) & 1;
  const int n0   = blockIdx.x * 64;
  const int m0   = blockIdx.y * 64;

  const float*  Af = (const float*)Av;
  const __bf16* Ab = (const __bf16*)Av;

  v8f acc[4] = {};

  const int mRow = wave * 16;
  const int kbA  = hi16 ? 16 : 0;
  const int kbB  = hi16 ? 16 : 0;

  for (int k0 = 0; k0 < DMODEL; k0 += 32) {
    // ---- stage A tile 64x32 ----
    if constexpr (A_BF16) {
      // bf16 source: async DMA straight to LDS (256 x 16B)
      #pragma unroll
      for (int i = 0; i < 2; ++i) {
        int t16 = i * 128 + tid;                 // 0..255
        int r = t16 >> 2, c8 = (t16 & 3) * 8;
        async_copy16(&sA[r][c8], Ab + (size_t)(m0 + r) * DMODEL + k0 + c8);
      }
    } else {
      // f32 source: vector load, convert, packed store
      #pragma unroll
      for (int i = 0; i < 4; ++i) {
        int t4 = i * 128 + tid;                  // 0..511 (f32x4 chunks)
        int r = t4 >> 3, c4 = (t4 & 7) * 4;
        f32x4 v4 = *(const f32x4*)&Af[(size_t)(m0 + r) * DMODEL + k0 + c4];
        bf16x4 b4;
        #pragma unroll
        for (int j = 0; j < 4; ++j) b4[j] = (__bf16)v4[j];
        *(bf16x4*)&sA[r][c4] = b4;
      }
    }
    // ---- stage W tile 32x64 transposed -> sB[n][k] ----
    #pragma unroll
    for (int i = 0; i < 4; ++i) {
      int t4 = i * 128 + tid;                    // 0..511 (f32x4 chunks)
      int kk = t4 >> 4, nn4 = (t4 & 15) * 4;
      f32x4 w4 = *(const f32x4*)&W[(size_t)(k0 + kk) * DMODEL + n0 + nn4];
      #pragma unroll
      for (int j = 0; j < 4; ++j) sB[nn4 + j][kk] = (__bf16)w4[j];
    }
    if (k0 + 32 < DMODEL)
      __builtin_prefetch(&W[(size_t)(k0 + 32 + (tid >> 6)) * DMODEL + n0 + (tid & 63)], 0, 0);
    if constexpr (A_BF16) async_wait();
    __syncthreads();

    v16bf a = frag_a(&sA[mRow + lh][0], kbA);
    #pragma unroll
    for (int ns = 0; ns < 4; ++ns) {
      v16bf b = frag_b(&sB[ns * 16 + lh][0], kbB);
      acc[ns] = wmma_bf16(a, b, acc[ns]);
    }
    __syncthreads();
  }

  // epilogue: C element (r, lane) -> (M = mRow + r + 8*hi16, N = ns*16 + lh)
  #pragma unroll
  for (int ns = 0; ns < 4; ++ns) {
    int n = n0 + ns * 16 + lh;
    float bv = bias[n];
    #pragma unroll
    for (int r = 0; r < 8; ++r) {
      int m = m0 + mRow + r + hi16 * 8;
      float c = (acc[ns][r] + bv) * scale;
      if (SPLIT_OUT) {
        int bb = m / SEQLEN, ll = m % SEQLEN;
        int hh = n >> 6, dd = n & 63;
        ((__bf16*)outv)[(((size_t)(bb * NUM_HEADS + hh)) * SEQLEN + ll) * HEAD_SIZE + dd] = (__bf16)c;
      } else {
        ((float*)outv)[(size_t)m * DMODEL + n] = c;
      }
    }
  }
}

// =====================================================================
// Flash attention: one block per (b, h, 64-query tile); 4 waves x 16 rows.
// Q and K tiles staged with async global->LDS DMA; V transposed manually.
// =====================================================================
__global__ __launch_bounds__(128) void attn_kernel(
    const __bf16* __restrict__ Q, const __bf16* __restrict__ K,
    const __bf16* __restrict__ V, const uint8_t* __restrict__ mask,
    __bf16* __restrict__ ctx)
{
  __shared__ __attribute__((aligned(16))) __bf16 sQ [64][72];  // [m][d]
  __shared__ __attribute__((aligned(16))) __bf16 sK [64][72];  // [key][d]  (= Bt for Q.K^T)
  __shared__ __attribute__((aligned(16))) __bf16 sVt[64][72];  // [d][key]  (= Bt for P.V)
  __shared__ __attribute__((aligned(16))) __bf16 sP [64][72];  // [m][key]
  __shared__ float sMask[64];

  const int tid  = threadIdx.x;
  const int lane = tid & 31;
  const int wave = tid >> 5;
  const int lh   = lane & 15;
  const int hi16 = (lane >> 4) & 1;
  const int qt = blockIdx.x;
  const int h  = blockIdx.y;
  const int b  = blockIdx.z;

  const size_t headBase = ((size_t)(b * NUM_HEADS + h)) * SEQLEN * HEAD_SIZE;
  const __bf16* Qb = Q + headBase + (size_t)qt * 64 * HEAD_SIZE;
  const __bf16* Kb = K + headBase;
  const __bf16* Vb = V + headBase;

  // Q tile: contiguous 64x64 bf16 = 512 x 16B, async DMA to LDS
  #pragma unroll
  for (int i = 0; i < 4; ++i) {
    int t16 = i * 128 + tid;                     // 0..511
    int r = t16 >> 3, c8 = (t16 & 7) * 8;
    async_copy16(&sQ[r][c8], Qb + t16 * 8);
  }

  const int mRow = wave * 16;
  const int kbA  = hi16 ? 16 : 0;
  const int kbB  = hi16 ? 16 : 0;

  v8f o[4] = {};
  float mSt[8], lSt[8];
  #pragma unroll
  for (int r = 0; r < 8; ++r) { mSt[r] = -3.0e38f; lSt[r] = 0.0f; }

  for (int kt = 0; kt < SEQLEN / 64; ++kt) {
    const __bf16* Kt = Kb + (size_t)kt * 64 * HEAD_SIZE;
    const __bf16* Vt = Vb + (size_t)kt * 64 * HEAD_SIZE;
    // K tile row-major: async DMA
    #pragma unroll
    for (int i = 0; i < 4; ++i) {
      int t16 = i * 128 + tid;
      async_copy16(&sK[t16 >> 3][(t16 & 7) * 8], Kt + t16 * 8);
    }
    // V tile transposed: vector load + scatter store
    #pragma unroll
    for (int i = 0; i < 4; ++i) {
      int t16 = i * 128 + tid;
      int r = t16 >> 3, c8 = (t16 & 7) * 8;
      bf16x8 vv = *(const bf16x8*)(Vt + t16 * 8);
      #pragma unroll
      for (int j = 0; j < 8; ++j) sVt[c8 + j][r] = vv[j];
    }
    if (tid < 64)
      sMask[tid] = mask[(size_t)b * SEQLEN + kt * 64 + tid] ? 0.0f : -1.0e9f;
    async_wait();
    __syncthreads();

    // S = Q . K^T  (d = 64 in two K-chunks of 32)
    v8f s[4] = {};
    #pragma unroll
    for (int dc = 0; dc < 64; dc += 32) {
      v16bf a = frag_a(&sQ[mRow + lh][dc], kbA);
      #pragma unroll
      for (int ns = 0; ns < 4; ++ns) {
        v16bf bb = frag_b(&sK[ns * 16 + lh][dc], kbB);
        s[ns] = wmma_bf16(a, bb, s[ns]);
      }
    }

    // online softmax; row stats replicated across each 16-lane half
    float alpha[8];
    #pragma unroll
    for (int r = 0; r < 8; ++r) {
      float rowmax = -3.0e38f;
      #pragma unroll
      for (int ns = 0; ns < 4; ++ns) {
        float v = s[ns][r] + sMask[ns * 16 + lh];
        s[ns][r] = v;
        rowmax = fmaxf(rowmax, v);
      }
      #pragma unroll
      for (int off = 8; off; off >>= 1)
        rowmax = fmaxf(rowmax, __shfl_xor(rowmax, off, 32));
      float mNew = fmaxf(mSt[r], rowmax);
      alpha[r] = __expf(mSt[r] - mNew);
      float rs = 0.0f;
      #pragma unroll
      for (int ns = 0; ns < 4; ++ns) {
        float p = __expf(s[ns][r] - mNew);
        s[ns][r] = p;
        rs += p;
      }
      #pragma unroll
      for (int off = 8; off; off >>= 1)
        rs += __shfl_xor(rs, off, 32);
      lSt[r] = lSt[r] * alpha[r] + rs;
      mSt[r] = mNew;
    }

    // spill P (C layout -> row-major LDS), rescale O
    #pragma unroll
    for (int ns = 0; ns < 4; ++ns)
      #pragma unroll
      for (int r = 0; r < 8; ++r)
        sP[mRow + r + hi16 * 8][ns * 16 + lh] = (__bf16)s[ns][r];
    #pragma unroll
    for (int ns = 0; ns < 4; ++ns)
      #pragma unroll
      for (int r = 0; r < 8; ++r)
        o[ns][r] *= alpha[r];
    __syncthreads();

    // O += P . V  (keys = 64 in two K-chunks of 32)
    #pragma unroll
    for (int kc = 0; kc < 64; kc += 32) {
      v16bf a = frag_a(&sP[mRow + lh][kc], kbA);
      #pragma unroll
      for (int ns = 0; ns < 4; ++ns) {
        v16bf bb = frag_b(&sVt[ns * 16 + lh][kc], kbB);
        o[ns] = wmma_bf16(a, bb, o[ns]);
      }
    }
    __syncthreads();
  }

  // normalize and write ctx [B][L][D] bf16 (merge heads)
  #pragma unroll
  for (int r = 0; r < 8; ++r) {
    float inv = lSt[r] > 0.0f ? 1.0f / lSt[r] : 0.0f;
    int lq = qt * 64 + mRow + r + hi16 * 8;
    #pragma unroll
    for (int ns = 0; ns < 4; ++ns) {
      int d = ns * 16 + lh;
      ctx[((size_t)b * SEQLEN + lq) * DMODEL + h * HEAD_SIZE + d] = (__bf16)(o[ns][r] * inv);
    }
  }
}

extern "C" void kernel_launch(void* const* d_in, const int* in_sizes, int n_in,
                              void* d_out, int out_size, void* d_ws, size_t ws_size,
                              hipStream_t stream) {
  const float*   k  = (const float*)d_in[0];
  const float*   v  = (const float*)d_in[1];
  const float*   q  = (const float*)d_in[2];
  const uint8_t* pm = (const uint8_t*)d_in[3];
  const float*   Wk = (const float*)d_in[4];
  const float*   bk = (const float*)d_in[5];
  const float*   Wv = (const float*)d_in[6];
  const float*   bv = (const float*)d_in[7];
  const float*   Wq = (const float*)d_in[8];
  const float*   bq = (const float*)d_in[9];
  const float*   Wo = (const float*)d_in[10];
  const float*   bo = (const float*)d_in[11];

  __bf16* ws = (__bf16*)d_ws;
  const size_t NPH = (size_t)BATCH * NUM_HEADS * SEQLEN * HEAD_SIZE; // 4Mi elems
  __bf16* Qh  = ws;
  __bf16* Kh  = ws + NPH;
  __bf16* Vh  = ws + 2 * NPH;
  __bf16* Ctx = ws + 3 * NPH;

  dim3 blk(128);
  dim3 gGemm(DMODEL / 64, (BATCH * SEQLEN) / 64);
  dim3 gAttn(SEQLEN / 64, NUM_HEADS, BATCH);

  const float qscale = 0.125f; // 1/sqrt(HEAD_SIZE)

  gemm_proj_kernel<false, true ><<<gGemm, blk, 0, stream>>>((const void*)k, Wk, bk, 1.0f,   (void*)Kh);
  gemm_proj_kernel<false, true ><<<gGemm, blk, 0, stream>>>((const void*)v, Wv, bv, 1.0f,   (void*)Vh);
  gemm_proj_kernel<false, true ><<<gGemm, blk, 0, stream>>>((const void*)q, Wq, bq, qscale, (void*)Qh);
  attn_kernel<<<gAttn, blk, 0, stream>>>(Qh, Kh, Vh, pm, Ctx);
  gemm_proj_kernel<true,  false><<<gGemm, blk, 0, stream>>>((const void*)Ctx, Wo, bo, 1.0f, d_out);
}
